// Hyper_73976516706864
// MI455X (gfx1250) — matrix operand order repaired and checked by reading, not compiled, not run on hardware
//
#include <hip/hip_runtime.h>
#include <hip/hip_bf16.h>

// ---------------------------------------------------------------------------
// Problem constants (from reference)
// ---------------------------------------------------------------------------
#define M_TOTAL   512          // C (num_chunks)
#define H_DIM     1024
#define X_DIM     96           // PE + CE
#define N_BIG     (7 * 16384)  // V * D = 114688

// GEMM tiling
#define BM    64
#define BN    128
#define KSTEP 32

typedef __attribute__((ext_vector_type(16))) __bf16 v16bf;
typedef __attribute__((ext_vector_type(8)))  float  v8f;
typedef __attribute__((ext_vector_type(4)))  __bf16 bf16x4;

union FragAB {
    v16bf v;
    uint4 u[2];
};

// Per-thread staging registers: A = 1 x uint4 (8 bf16), B = 4 x float4.
struct StageRegs {
    uint4  a;
    float4 b[4];
};

// ---------------------------------------------------------------------------
// Kernel 0: build x = concat(preference @ pref_emb, chunk_emb)  -> bf16 [512,96]
// ---------------------------------------------------------------------------
__global__ void build_x_kernel(const float* __restrict__ pref,
                               const float* __restrict__ pref_emb,   // [2,32]
                               const float* __restrict__ chunk_emb,  // [512,64]
                               __bf16* __restrict__ x) {             // [512,96]
    int idx = blockIdx.x * blockDim.x + threadIdx.x;   // 512*96 threads
    int c = idx / X_DIM;
    int j = idx - c * X_DIM;
    float v;
    if (j < 32) {
        v = pref[0] * pref_emb[j] + pref[1] * pref_emb[32 + j];
    } else {
        v = chunk_emb[c * 64 + (j - 32)];
    }
    x[idx] = (__bf16)v;
}

// ---------------------------------------------------------------------------
// Staging helpers (256 threads/block).
//   Global -> regs: issued as one clause; latency hidden across WMMA phase.
//   Regs -> LDS:    fp32 -> bf16 conversion happens here (packed cvt).
// Both A and B tiles are stored k-contiguous, row stride 64 bytes.
// ---------------------------------------------------------------------------
template <int K_TOTAL>
__device__ __forceinline__ void stage_load(const __bf16* __restrict__ A,
                                           const float*  __restrict__ B,
                                           int mBlk, int nBlk, int k0, int tid,
                                           StageRegs& r) {
    // A: 64 rows x 32 k bf16 = 256 x uint4, one per thread
    {
        const int row = tid >> 2;
        const int q   = tid & 3;                      // 4 x 16B per row
        r.a = *(const uint4*)(A + (size_t)(mBlk + row) * K_TOTAL + k0 + q * 8);
    }
    // B: 128 rows x 32 k f32 = 1024 x float4, four per thread
#pragma unroll
    for (int i = 0; i < 4; ++i) {
        const int idx = tid + i * 256;
        const int row = idx >> 3;
        const int q   = idx & 7;                      // 8 x float4 per row
        r.b[i] = *(const float4*)(B + (size_t)(nBlk + row) * K_TOTAL + k0 + q * 4);
    }
}

__device__ __forceinline__ void stage_store(unsigned char* __restrict__ lsA,
                                            unsigned char* __restrict__ lsB,
                                            int tid, const StageRegs& r) {
    {
        const int row = tid >> 2;
        const int q   = tid & 3;
        *(uint4*)(lsA + row * 64 + q * 16) = r.a;
    }
#pragma unroll
    for (int i = 0; i < 4; ++i) {
        const int idx = tid + i * 256;
        const int row = idx >> 3;
        const int q   = idx & 7;
        bf16x4 h;
        h.x = (__bf16)r.b[i].x; h.y = (__bf16)r.b[i].y;
        h.z = (__bf16)r.b[i].z; h.w = (__bf16)r.b[i].w;
        *(bf16x4*)(lsB + row * 64 + q * 8) = h;
    }
}

// ---------------------------------------------------------------------------
// Templated WMMA GEMM:  out[m,n] = act( sum_k A[m,k] * B[n,k] + bias[n] )
//   A: bf16 [512, K_TOTAL] row-major (scratch)
//   B: f32  [N_TOTAL, K_TOTAL] row-major (global; converted to bf16 on stage)
//   out: bf16 (scratch) or f32 (d_out, non-temporal)
// Block: 256 threads = 8 waves; tile 64(M) x 128(N); wave = 2x2 16x16 tiles.
// Double-buffered LDS, register prefetch, one barrier per K-step.
// Grid is 1D; M-block varies fastest so the 8 blocks sharing a B tile are
// consecutive (L2 reuse of ws).
// ---------------------------------------------------------------------------
template <int K_TOTAL, bool HAS_BIAS, bool RELU, bool BF16_OUT>
__global__ __launch_bounds__(256)
void gemm_wmma_kernel(const __bf16* __restrict__ A,
                      const float*  __restrict__ B,
                      const float*  __restrict__ bias,
                      void* __restrict__ out,
                      int N_TOTAL) {
    // Double-buffered: A tile 4 KB x2, B tile 8 KB x2 = 24 KB LDS.
    __shared__ __align__(16) unsigned char lsA[2][BM * KSTEP * 2];
    __shared__ __align__(16) unsigned char lsB[2][BN * KSTEP * 2];

    const int nTilesM = M_TOTAL / BM;                 // 8
    const int bid  = blockIdx.x;
    const int mBlk = (bid & (nTilesM - 1)) * BM;      // M fastest -> L2 reuse of B
    const int nBlk = (bid / nTilesM) * BN;

    const int tid   = threadIdx.x;
    const int lane  = tid & 31;
    const int wave  = tid >> 5;
    const int waveM = wave >> 2;                      // 0..1  (32 rows each)
    const int waveN = wave & 3;                       // 0..3  (32 cols each)
    const int l15   = lane & 15;
    const int khalf = lane >> 4;                      // lane-half selects K subset

    v8f acc[2][2] = {};

    float biasv[2] = {0.0f, 0.0f};
    if (HAS_BIAS) {
#pragma unroll
        for (int nt = 0; nt < 2; ++nt)
            biasv[nt] = bias[nBlk + waveN * 32 + nt * 16 + l15];
    }

    // Precomputed LDS fragment base offsets (constant across K-steps).
    const int offA = (waveM * 32 + l15) * 64 + khalf * 16;   // + mt*16*64
    const int offB = (waveN * 32 + l15) * 64 + khalf * 16;   // + nt*16*64

    constexpr int nK = K_TOTAL / KSTEP;
    StageRegs sr;
    stage_load<K_TOTAL>(A, B, mBlk, nBlk, 0, tid, sr);

    for (int kk = 0; kk < nK; ++kk) {
        const int buf = kk & 1;
        unsigned char* la = lsA[buf];
        unsigned char* lb = lsB[buf];

        // Drain prefetch regs into this K-step's LDS buffer (cvt to bf16).
        stage_store(la, lb, tid, sr);
        // Prefetch next K-step from global into regs (one load clause; its
        // latency is covered by the WMMA phase below).
        if (kk + 1 < nK)
            stage_load<K_TOTAL>(A, B, mBlk, nBlk, (kk + 1) * KSTEP, tid, sr);

        __syncthreads();   // stores of `buf` visible; prior reads of `buf` done

        // ---- fragment loads (ISA 7.12.2 16-bit layouts) --------------------
        FragAB a[2], b[2];
#pragma unroll
        for (int mt = 0; mt < 2; ++mt) {
            const unsigned char* p = la + offA + mt * 16 * 64;
            a[mt].u[0] = *(const uint4*)(p);
            a[mt].u[1] = *(const uint4*)(p + 32);
        }
#pragma unroll
        for (int nt = 0; nt < 2; ++nt) {
            const unsigned char* p = lb + offB + nt * 16 * 64;
            b[nt].u[0] = *(const uint4*)(p);
            b[nt].u[1] = *(const uint4*)(p + 32);
        }

#pragma unroll
        for (int mt = 0; mt < 2; ++mt)
#pragma unroll
            for (int nt = 0; nt < 2; ++nt)
                acc[mt][nt] = __builtin_amdgcn_wmma_f32_16x16x32_bf16(
                    false, a[mt].v, false, b[nt].v,
                    (short)0, acc[mt][nt], false, false);
    }

    // ---- write out: C/D layout VGPR r -> M = r + 8*khalf, N = l15 ----------
#pragma unroll
    for (int mt = 0; mt < 2; ++mt) {
#pragma unroll
        for (int nt = 0; nt < 2; ++nt) {
            const int n     = nBlk + waveN * 32 + nt * 16 + l15;
            const int mbase = mBlk + waveM * 32 + mt * 16 + khalf * 8;
#pragma unroll
            for (int r = 0; r < 8; ++r) {
                float v = acc[mt][nt][r];
                if (HAS_BIAS) v += biasv[nt];
                if (RELU)     v = fmaxf(v, 0.0f);
                const size_t off = (size_t)(mbase + r) * (size_t)N_TOTAL + n;
                if (BF16_OUT) {
                    ((__bf16*)out)[off] = (__bf16)v;
                } else {
                    __builtin_nontemporal_store(v, (float*)out + off);
                }
            }
        }
    }
}

// ---------------------------------------------------------------------------
// Launch
// ---------------------------------------------------------------------------
extern "C" void kernel_launch(void* const* d_in, const int* in_sizes, int n_in,
                              void* d_out, int out_size, void* d_ws, size_t ws_size,
                              hipStream_t stream) {
    const float* pref   = (const float*)d_in[0];   // [2]
    const float* prefE  = (const float*)d_in[1];   // [2,32]
    const float* chunkE = (const float*)d_in[2];   // [512,64]
    const float* W1     = (const float*)d_in[3];   // [1024,96]
    const float* b1     = (const float*)d_in[4];   // [1024]
    const float* W2     = (const float*)d_in[5];   // [1024,1024]
    const float* b2     = (const float*)d_in[6];
    const float* W3     = (const float*)d_in[7];
    const float* b3     = (const float*)d_in[8];
    const float* ws     = (const float*)d_in[9];   // [7,16384,1024] == [114688,1024]
    float* out = (float*)d_out;                    // [512 * 114688] f32

    // Scratch layout (bf16): x 96KB | h1 1MB | h2 1MB | rep 1MB
    unsigned char* sc = (unsigned char*)d_ws;
    __bf16* x   = (__bf16*)(sc);
    __bf16* h1  = (__bf16*)(sc + (128u << 10));
    __bf16* h2  = (__bf16*)(sc + (128u << 10) + (1u << 20));
    __bf16* rep = (__bf16*)(sc + (128u << 10) + (2u << 20));

    // 0) build bf16 input features
    build_x_kernel<<<dim3((M_TOTAL * X_DIM) / 256), dim3(256), 0, stream>>>(
        pref, prefE, chunkE, x);

    const int gM = M_TOTAL / BM;                       // 8
    // 1) h1 = relu(x @ W1^T + b1)        [512,96] -> [512,1024]
    gemm_wmma_kernel<X_DIM, true, true, true>
        <<<dim3(gM * (H_DIM / BN)), dim3(256), 0, stream>>>(x, W1, b1, h1, H_DIM);
    // 2) h2 = relu(h1 @ W2^T + b2)
    gemm_wmma_kernel<H_DIM, true, true, true>
        <<<dim3(gM * (H_DIM / BN)), dim3(256), 0, stream>>>(h1, W2, b2, h2, H_DIM);
    // 3) rep = h2 @ W3^T + b3            (linear)
    gemm_wmma_kernel<H_DIM, true, false, true>
        <<<dim3(gM * (H_DIM / BN)), dim3(256), 0, stream>>>(h2, W3, b3, rep, H_DIM);
    // 4) out = rep @ ws^T                M=512, N=114688, K=1024 (120 GFLOP)
    gemm_wmma_kernel<H_DIM, false, false, false>
        <<<dim3(gM * (N_BIG / BN)), dim3(256), 0, stream>>>(rep, ws, nullptr, out, N_BIG);
}
